// LEIGNN_34376918238025
// MI455X (gfx1250) — matrix operand order repaired and checked by reading, not compiled, not run on hardware
//
#include <hip/hip_runtime.h>
#include <math.h>

// ---------------- problem constants ----------------
#define N_NODES 5000
#define N_EDGES 100000
#define H       256
#define H2      128
#define H3      768
#define NRBF    128
#define NLAYERS 2
#define NBATCH  16

#define INV3_C 0.5773502691896258f   // 1/sqrt(3)
#define INVH_C 0.0625f               // 1/sqrt(256)
#define INV2_C 0.7071067811865475f   // 1/sqrt(2)

typedef __attribute__((ext_vector_type(16))) _Float16 v16h;
typedef __attribute__((ext_vector_type(8)))  _Float16 v8h;
typedef __attribute__((ext_vector_type(8)))  float    v8f;

__device__ __forceinline__ float ssilu_f(float v) {
  // silu(v)/0.6 with fast v_rcp_f32 instead of IEEE divide
  return v * __builtin_amdgcn_rcpf(1.f + __expf(-v)) * (1.f / 0.6f);
}

// ---------------- small utility kernels ----------------
__global__ void fill0_k(float* __restrict__ p, int n) {
  int i = blockIdx.x * 256 + threadIdx.x;
  if (i < n) p[i] = 0.f;
}

__global__ void init_x_k(const int* __restrict__ z, const float* __restrict__ emb,
                         float* __restrict__ x) {
  int idx = blockIdx.x * 256 + threadIdx.x;
  if (idx >= N_NODES * H) return;
  int n = idx >> 8, h = idx & 255;
  x[idx] = emb[(size_t)z[n] * H + h];
}

// ---------------- weight conversion into WMMA fragment-swizzled f16 ----------------
// Panel = 32(K) x 16(N) tile = 512 halves, stored so that lane L's 16 B-fragment
// halves are contiguous:  dst[p*512 + L*16 + t] = src[(kTile*32 + (L&16)/2 + (t&7)
//                                                     + (t&8)*2)*Nn + nTile*16 + (L&15)]
// Panel order: p = nTile * kTiles + kTile  (k-panels contiguous per column tile).
__global__ void cvt_swz_k(const float* __restrict__ src, _Float16* __restrict__ dst,
                          int Nn, int kTilesLog, int total) {
  int idx = blockIdx.x * 256 + threadIdx.x;
  if (idx >= total) return;
  int p = idx >> 9, within = idx & 511;
  int lane = within >> 4, t = within & 15;
  int kTile = p & ((1 << kTilesLog) - 1);
  int nTile = p >> kTilesLog;
  int k = (kTile << 5) + ((lane & 16) >> 1) + (t & 7) + ((t & 8) << 1);
  int n = (nTile << 4) + (lane & 15);
  dst[idx] = (_Float16)src[(size_t)k * Nn + n];
}

// ---------------- edge geometry + radial basis (f16 output) ----------------
__global__ void edge_geom_k(const float* __restrict__ pos, const int* __restrict__ eidx,
                            _Float16* __restrict__ rbf, float* __restrict__ ev) {
  int e = blockIdx.x * 256 + threadIdx.x;
  if (e >= N_EDGES) return;
  int j = eidx[e], i = eidx[N_EDGES + e];
  float dx = pos[j * 3 + 0] - pos[i * 3 + 0];
  float dy = pos[j * 3 + 1] - pos[i * 3 + 1];
  float dz = pos[j * 3 + 2] - pos[i * 3 + 2];
  float dist = sqrtf(dx * dx + dy * dy + dz * dz);
  float inv = __builtin_amdgcn_rcpf(dist);
  ev[e * 3 + 0] = -dx * inv;
  ev[e * 3 + 1] = -dy * inv;
  ev[e * 3 + 2] = -dz * inv;
  float d = dist * (1.f / 6.0f);            // d / cutoff
  float d2 = d * d, d4 = d2 * d2, d5 = d4 * d;
  float env = 1.f - 21.f * d5 + 35.f * d5 * d - 15.f * d5 * d2;  // p=5 envelope
  if (d >= 1.f) env = 0.f;
  const float delta = 1.f / (NRBF - 1);
  const float coef = -0.5f / (delta * delta);
  for (int r = 0; r < NRBF; ++r) {
    float diff = d - r * delta;
    rbf[(size_t)e * NRBF + r] = (_Float16)(env * __expf(coef * diff * diff));
  }
}

// Combine two aligned v8h LDS loads into a WMMA A fragment.
__device__ __forceinline__ v16h load_a_frag(const _Float16* ap, int k0, int hi8) {
  v8h alo = *(const v8h*)(ap + k0 + hi8);       // K = k0+hi8   .. +7  (contiguous)
  v8h ahi = *(const v8h*)(ap + k0 + hi8 + 16);  // K = k0+hi8+16.. +23 (contiguous)
  return __builtin_shufflevector(alo, ahi, 0, 1, 2, 3, 4, 5, 6, 7,
                                 8, 9, 10, 11, 12, 13, 14, 15);
}

// ---------------- generic WMMA GEMM: C = act(A[f32] x Bswz[f16] + bias) ----------------
// grid = (ceil(M/16), Nn/128); block = 256 (8 waves, each owns one 16-wide col tile)
__global__ void gemm_k(const float* __restrict__ A, const _Float16* __restrict__ Bw,
                       const float* __restrict__ bias, float* __restrict__ C,
                       int M, int Nn, int kLog, int act) {
  __shared__ __align__(16) _Float16 aLds[16 * 512];
  const int K = 1 << kLog;
  const int kTiles = K >> 5;
  const int mBase = blockIdx.x * 16;
  const int nBase = blockIdx.y * 128;
  const int tid = threadIdx.x;
  for (int idx = tid; idx < (16 << kLog); idx += 256) {
    int m = idx >> kLog, k = idx & (K - 1);
    float v = 0.f;
    if (mBase + m < M) v = A[(size_t)(mBase + m) * K + k];
    aLds[idx] = (_Float16)v;
  }
  __syncthreads();
  const int lane = tid & 31, wave = tid >> 5;
  const int nlo = lane & 15, hi8 = (lane & 16) ? 8 : 0;
  const int col = nBase + wave * 16 + nlo;
  const _Float16* ap = aLds + (nlo << kLog);
  const _Float16* bp = Bw + ((size_t)(col >> 4) * kTiles) * 512 + lane * 16;
  v8f acc = {};
  for (int kt = 0; kt < kTiles; ++kt) {
    v16h a = load_a_frag(ap, kt << 5, hi8);
    v16h b = *(const v16h*)(bp + (size_t)kt * 512);
    acc = __builtin_amdgcn_wmma_f32_16x16x32_f16(false, a, false, b, (short)0, acc,
                                                 false, false);
  }
#pragma unroll
  for (int r = 0; r < 8; ++r) {
    int m = mBase + r + (hi8 ? 8 : 0);
    if (m < M) {
      float v = acc[r];
      if (bias) v += bias[col];
      if (act) v = ssilu_f(v);
      C[(size_t)m * Nn + col] = v;
    }
  }
}

// ---------------- fused edge message: WMMA rbf GEMM + gather + scatter ----------------
// block = 256 threads handles 16 edges; rbf_h tile (16x768) never hits HBM.
// Wr is fragment-swizzled [128,768] (kTiles = 4, nTiles = 48).
__global__ void edge_msg_k(const _Float16* __restrict__ rbf,   // [E,128] f16
                           const _Float16* __restrict__ Wr,    // swizzled, L2-hot
                           const float* __restrict__ br,       // [768]
                           const float* __restrict__ xh,       // [N,768]
                           const float* __restrict__ vec,      // [N,3,256]
                           const float* __restrict__ ev,       // [E,3]
                           const int* __restrict__ ejp, const int* __restrict__ eip,
                           float* __restrict__ dvec, float* __restrict__ dxp) {
  __shared__ __align__(16) _Float16 aLds[16 * NRBF];  // 4 KB
  __shared__ float rh[16 * H3];                        // 48 KB
  const int e0 = blockIdx.x * 16;
  const int tid = threadIdx.x;
  for (int idx = tid; idx < 16 * NRBF; idx += 256) {
    int e = idx >> 7, k = idx & 127;
    _Float16 v = (_Float16)0.f;
    if (e0 + e < N_EDGES) v = rbf[(size_t)(e0 + e) * NRBF + k];
    aLds[idx] = v;
  }
  __syncthreads();
  const int lane = tid & 31, wave = tid >> 5;
  const int nlo = lane & 15, hi8 = (lane & 16) ? 8 : 0;
  const _Float16* ap = aLds + nlo * NRBF;
  // column tiles wave*6 .. wave*6+5 ; panel index = tile*4 + kt
  const _Float16* bp = Wr + (size_t)(wave * 6) * 4 * 512 + lane * 16;
  v8f zero = {};
  v8f acc[6];
#pragma unroll
  for (int t = 0; t < 6; ++t) acc[t] = zero;
  for (int kt = 0; kt < 4; ++kt) {
    v16h a = load_a_frag(ap, kt << 5, hi8);
#pragma unroll
    for (int t = 0; t < 6; ++t) {
      v16h b = *(const v16h*)(bp + ((size_t)t * 4 + kt) * 512);
      acc[t] = __builtin_amdgcn_wmma_f32_16x16x32_f16(false, a, false, b, (short)0,
                                                      acc[t], false, false);
    }
  }
#pragma unroll
  for (int t = 0; t < 6; ++t) {
    int col = wave * 96 + t * 16 + nlo;
#pragma unroll
    for (int r = 0; r < 8; ++r) {
      int m = r + (hi8 ? 8 : 0);
      rh[m * H3 + col] = acc[t][r] + br[col];
    }
  }
  __syncthreads();
  // epilogue: thread h handles channel h for all 16 edges; coalesced gathers per edge
  const int h = tid;  // blockDim.x == 256 == H
  for (int e = 0; e < 16; ++e) {
    int ge = e0 + e;
    if (ge >= N_EDGES) break;
    int jn = ejp[ge], in = eip[ge];
    float m1 = xh[(size_t)jn * H3 + h]       * rh[e * H3 + h]       * INV3_C;
    float m2 = xh[(size_t)jn * H3 + 256 + h] * rh[e * H3 + 256 + h] * INV3_C;
    float m3 = xh[(size_t)jn * H3 + 512 + h] * rh[e * H3 + 512 + h] * INV3_C;
#pragma unroll
    for (int s = 0; s < 3; ++s) {
      float vji = (m1 * vec[(size_t)(jn * 3 + s) * H + h] + m2 * ev[ge * 3 + s]) * INVH_C;
      atomicAdd(&dvec[(size_t)(in * 3 + s) * H + h], vji);
    }
    atomicAdd(&dxp[(size_t)in * H + h], m3);
  }
}

// ---------------- residual / update elementwise kernels ----------------
__global__ void resid_x_k(float* __restrict__ x, const float* __restrict__ d, int n) {
  int i = blockIdx.x * 256 + threadIdx.x;
  if (i < n) x[i] = (x[i] + d[i]) * INV2_C;
}

__global__ void addv_k(float* __restrict__ a, const float* __restrict__ b, int n) {
  int i = blockIdx.x * 256 + threadIdx.x;
  if (i < n) a[i] += b[i];
}

__global__ void upd_prep_k(const float* __restrict__ vp, const float* __restrict__ x,
                           float* __restrict__ xin, float* __restrict__ vdot) {
  int idx = blockIdx.x * 256 + threadIdx.x;
  if (idx >= N_NODES * H) return;
  int n = idx >> 8, h = idx & 255;
  float dsum = 0.f, nsum = 0.f;
#pragma unroll
  for (int s = 0; s < 3; ++s) {
    float v1 = vp[(size_t)(n * 3 + s) * 512 + h];
    float v2 = vp[(size_t)(n * 3 + s) * 512 + 256 + h];
    dsum += v1 * v2;
    nsum += v2 * v2;
  }
  vdot[idx] = dsum * INVH_C;
  xin[(size_t)n * 512 + h]       = x[idx];
  xin[(size_t)n * 512 + 256 + h] = sqrtf(nsum + 1e-8f);
}

__global__ void upd_apply_k(const float* __restrict__ xh, const float* __restrict__ vdot,
                            const float* __restrict__ vp, float* __restrict__ x,
                            float* __restrict__ vec) {
  int idx = blockIdx.x * 256 + threadIdx.x;
  if (idx >= N_NODES * H) return;
  int n = idx >> 8, h = idx & 255;
  float xv1 = xh[(size_t)n * H3 + h];
  float xv2 = xh[(size_t)n * H3 + 256 + h];
  float xv3 = xh[(size_t)n * H3 + 512 + h];
  x[idx] += (xv1 + xv2 * vdot[idx]) * INV2_C;
#pragma unroll
  for (int s = 0; s < 3; ++s)
    vec[(size_t)(n * 3 + s) * H + h] += xv3 * vp[(size_t)(n * 3 + s) * 512 + h];
}

// ---------------- readout ----------------
__global__ void energy_k(const float* __restrict__ h1, const float* __restrict__ oeW2,
                         const float* __restrict__ oeb2, const int* __restrict__ batch,
                         float* __restrict__ out) {
  int n = blockIdx.x * 256 + threadIdx.x;
  if (n >= N_NODES) return;
  float pe = oeb2[0];
  for (int k = 0; k < H2; ++k) pe += h1[(size_t)n * H2 + k] * oeW2[k];
  atomicAdd(&out[batch[n]], pe);
}

__global__ void forces_k(const float* __restrict__ vec, const float* __restrict__ ofW,
                         float* __restrict__ out) {
  int idx = blockIdx.x * 256 + threadIdx.x;  // n*3+s
  if (idx >= N_NODES * 3) return;
  float f = 0.f;
  for (int h = 0; h < H; ++h) f += vec[(size_t)idx * H + h] * ofW[h];
  out[idx] = f;
}

// ---------------- host driver ----------------
extern "C" void kernel_launch(void* const* d_in, const int* in_sizes, int n_in,
                              void* d_out, int out_size, void* d_ws, size_t ws_size,
                              hipStream_t stream) {
  (void)in_sizes; (void)n_in; (void)out_size; (void)ws_size;
  const float* pos      = (const float*)d_in[0];
  const int*   z        = (const int*)d_in[1];
  const int*   eidx     = (const int*)d_in[2];
  const int*   batch    = (const int*)d_in[3];
  const float* emb      = (const float*)d_in[4];
  const float* msg_xW1  = (const float*)d_in[5];
  const float* msg_xb1  = (const float*)d_in[6];
  const float* msg_xW2  = (const float*)d_in[7];
  const float* msg_xb2  = (const float*)d_in[8];
  const float* msg_rbfW = (const float*)d_in[9];
  const float* msg_rbfb = (const float*)d_in[10];
  const float* upd_vecW = (const float*)d_in[11];
  const float* upd_xW1  = (const float*)d_in[12];
  const float* upd_xb1  = (const float*)d_in[13];
  const float* upd_xW2  = (const float*)d_in[14];
  const float* upd_xb2  = (const float*)d_in[15];
  const float* oeW1     = (const float*)d_in[16];
  const float* oeb1     = (const float*)d_in[17];
  const float* oeW2     = (const float*)d_in[18];
  const float* oeb2     = (const float*)d_in[19];
  const float* ofW      = (const float*)d_in[20];
  float* out = (float*)d_out;

  char* ws = (char*)d_ws;
  size_t off = 0;
  auto alloc = [&](size_t bytes) -> void* {
    void* p = ws + off;
    off = (off + bytes + 255) & ~(size_t)255;
    return p;
  };
  float* x    = (float*)alloc((size_t)N_NODES * H * 4);
  float* vec  = (float*)alloc((size_t)N_NODES * 3 * H * 4);
  float* dxb  = (float*)alloc((size_t)N_NODES * H * 4);
  float* dvec = (float*)alloc((size_t)N_NODES * 3 * H * 4);
  float* xh   = (float*)alloc((size_t)N_NODES * H3 * 4);
  float* h1   = (float*)alloc((size_t)N_NODES * H2 * 4);
  float* h256 = (float*)alloc((size_t)N_NODES * H * 4);
  float* vp   = (float*)alloc((size_t)N_NODES * 3 * 2 * H * 4);
  float* xin  = (float*)alloc((size_t)N_NODES * 2 * H * 4);
  float* vdot = (float*)alloc((size_t)N_NODES * H * 4);
  float* ev   = (float*)alloc((size_t)N_EDGES * 3 * 4);
  _Float16* rbf16      = (_Float16*)alloc((size_t)N_EDGES * NRBF * 2);
  _Float16* w_msg_xW1  = (_Float16*)alloc((size_t)NLAYERS * H * H2 * 2);
  _Float16* w_msg_xW2  = (_Float16*)alloc((size_t)NLAYERS * H2 * H3 * 2);
  _Float16* w_msg_rbfW = (_Float16*)alloc((size_t)NLAYERS * NRBF * H3 * 2);
  _Float16* w_upd_vecW = (_Float16*)alloc((size_t)NLAYERS * H * 2 * H * 2);
  _Float16* w_upd_xW1  = (_Float16*)alloc((size_t)NLAYERS * 2 * H * H * 2);
  _Float16* w_upd_xW2  = (_Float16*)alloc((size_t)NLAYERS * H * H3 * 2);
  _Float16* w_oeW1     = (_Float16*)alloc((size_t)H * H2 * 2);

  // fragment-swizzled weight conversion; kTilesLog = log2(K/32)
  auto cvts = [&](const float* s, _Float16* d, int K, int Nn, int kTilesLog) {
    int n = K * Nn;
    cvt_swz_k<<<dim3((n + 255) / 256), dim3(256), 0, stream>>>(s, d, Nn, kTilesLog, n);
  };
  for (int l = 0; l < NLAYERS; ++l) {
    cvts(msg_xW1  + (size_t)l * H * H2,    w_msg_xW1  + (size_t)l * H * H2,    H,     H2,    3);
    cvts(msg_xW2  + (size_t)l * H2 * H3,   w_msg_xW2  + (size_t)l * H2 * H3,   H2,    H3,    2);
    cvts(msg_rbfW + (size_t)l * NRBF * H3, w_msg_rbfW + (size_t)l * NRBF * H3, NRBF,  H3,    2);
    cvts(upd_vecW + (size_t)l * H * 2 * H, w_upd_vecW + (size_t)l * H * 2 * H, H,     2 * H, 3);
    cvts(upd_xW1  + (size_t)l * 2 * H * H, w_upd_xW1  + (size_t)l * 2 * H * H, 2 * H, H,     4);
    cvts(upd_xW2  + (size_t)l * H * H3,    w_upd_xW2  + (size_t)l * H * H3,    H,     H3,    3);
  }
  cvts(oeW1, w_oeW1, H, H2, 3);

  auto fill0 = [&](float* p, int n) {
    fill0_k<<<dim3((n + 255) / 256), dim3(256), 0, stream>>>(p, n);
  };
  auto gemm = [&](const float* A, const _Float16* Bw, const float* bias, float* C,
                  int M, int Nn, int kLog, int act) {
    dim3 g((M + 15) / 16, Nn / 128);
    gemm_k<<<g, dim3(256), 0, stream>>>(A, Bw, bias, C, M, Nn, kLog, act);
  };

  init_x_k<<<dim3((N_NODES * H + 255) / 256), dim3(256), 0, stream>>>(z, emb, x);
  fill0(vec, N_NODES * 3 * H);
  edge_geom_k<<<dim3((N_EDGES + 255) / 256), dim3(256), 0, stream>>>(pos, eidx, rbf16, ev);

  for (int l = 0; l < NLAYERS; ++l) {
    fill0(dxb, N_NODES * H);
    fill0(dvec, N_NODES * 3 * H);
    // message MLP on nodes
    gemm(x,  w_msg_xW1 + (size_t)l * H * H2,  msg_xb1 + l * H2, h1, N_NODES, H2, 8, 1);
    gemm(h1, w_msg_xW2 + (size_t)l * H2 * H3, msg_xb2 + l * H3, xh, N_NODES, H3, 7, 0);
    // fused rbf GEMM + gather + scatter
    edge_msg_k<<<dim3((N_EDGES + 15) / 16), dim3(256), 0, stream>>>(
        rbf16, w_msg_rbfW + (size_t)l * NRBF * H3, msg_rbfb + l * H3, xh, vec, ev,
        eidx, eidx + N_EDGES, dvec, dxb);
    resid_x_k<<<dim3((N_NODES * H + 255) / 256), dim3(256), 0, stream>>>(x, dxb,
                                                                         N_NODES * H);
    addv_k<<<dim3((N_NODES * 3 * H + 255) / 256), dim3(256), 0, stream>>>(
        vec, dvec, N_NODES * 3 * H);
    // update
    gemm(vec, w_upd_vecW + (size_t)l * H * 2 * H, nullptr, vp, 3 * N_NODES, 2 * H, 8, 0);
    upd_prep_k<<<dim3((N_NODES * H + 255) / 256), dim3(256), 0, stream>>>(vp, x, xin,
                                                                          vdot);
    gemm(xin,  w_upd_xW1 + (size_t)l * 2 * H * H, upd_xb1 + l * H,  h256, N_NODES, H, 9, 1);
    gemm(h256, w_upd_xW2 + (size_t)l * H * H3,    upd_xb2 + l * H3, xh,   N_NODES, H3, 8, 0);
    upd_apply_k<<<dim3((N_NODES * H + 255) / 256), dim3(256), 0, stream>>>(xh, vdot, vp,
                                                                           x, vec);
  }

  // readout
  gemm(x, w_oeW1, oeb1, h1, N_NODES, H2, 8, 1);
  fill0(out, NBATCH);
  energy_k<<<dim3((N_NODES + 255) / 256), dim3(256), 0, stream>>>(h1, oeW2, oeb2, batch,
                                                                  out);
  forces_k<<<dim3((N_NODES * 3 + 255) / 256), dim3(256), 0, stream>>>(vec, ofW,
                                                                      out + NBATCH);
}
